// Sae_79336635892535
// MI455X (gfx1250) — matrix operand order repaired and checked by reading, not compiled, not run on hardware
//
#include <hip/hip_runtime.h>
#include <hip/hip_bf16.h>
#include <stdint.h>

#define D_IN        2048
#define NUM_LATENTS 32768
#define TOPK        64
#define NROWS       4096

// ---------------- WMMA types ----------------
typedef __attribute__((ext_vector_type(16))) __bf16 v16bf;
typedef __attribute__((ext_vector_type(8)))  float  v8f;
typedef __attribute__((ext_vector_type(4)))  int    v4i;

union FragBF16 {
    v16bf v;
    uint4 q[2];   // two 16-byte chunks = 16 bf16
};

// pointer types matching the async-to-LDS builtin: (v4i AS1*, v4i AS3*, imm, imm)
typedef __attribute__((address_space(1))) v4i* gptr_v4i;
typedef __attribute__((address_space(3))) v4i* lptr_v4i;

__device__ __forceinline__ void async_ld_b128(const void* g, void* l) {
    __builtin_amdgcn_global_load_async_to_lds_b128((gptr_v4i)g, (lptr_v4i)l, 0, 0);
}

// ---------------- fp32 -> bf16 (RNE) ----------------
__device__ __forceinline__ unsigned short f2bf(float f) {
    unsigned int u = __float_as_uint(f);
    unsigned int r = u + 0x7FFFu + ((u >> 16) & 1u);
    return (unsigned short)(r >> 16);
}

__global__ __launch_bounds__(256) void cvt_f32_bf16(const float* __restrict__ s,
                                                    unsigned short* __restrict__ d,
                                                    int n4) {
    int i = blockIdx.x * blockDim.x + threadIdx.x;
    if (i < n4) {
        float4 f = ((const float4*)s)[i];
        ushort4 o;
        o.x = f2bf(f.x); o.y = f2bf(f.y); o.z = f2bf(f.z); o.w = f2bf(f.w);
        ((ushort4*)d)[i] = o;
    }
}

// ---------------- Encoder GEMM: pre[n][l] = sum_d x[n][d]*W[l][d] + b_enc[l] ----------------
// Block tile 128(M) x 256(N), BK=32, 256 threads = 8 waves in 2(M) x 4(N) grid.
// Each wave: 64x64 = 4x4 WMMA tiles (16 accumulators).
// Double-buffered LDS filled by GLOBAL_LOAD_ASYNC_TO_LDS_B128 (ASYNCcnt).
#define BM 128
#define BN 256
#define BK 32
#define LDSS 40   // padded row stride in bf16 elems (80B = 20 dwords -> conflict-free)

// per-thread async chunks per K-step: A = 128rows*4/256 = 2, B = 256rows*4/256 = 4
__device__ __forceinline__ void load_tiles_async(
    const unsigned short* __restrict__ Xg, const unsigned short* __restrict__ Wg,
    unsigned short* Abuf, unsigned short* Bbuf,
    int tid, int block_m, int block_n, int k0)
{
    #pragma unroll
    for (int it = 0; it < 2; ++it) {
        int id = it * 256 + tid;
        int r  = id >> 2;
        int c  = (id & 3) * 8;
        async_ld_b128(Xg + (size_t)(block_m + r) * D_IN + k0 + c, Abuf + r * LDSS + c);
    }
    #pragma unroll
    for (int it = 0; it < 4; ++it) {
        int id = it * 256 + tid;
        int r  = id >> 2;
        int c  = (id & 3) * 8;
        async_ld_b128(Wg + (size_t)(block_n + r) * D_IN + k0 + c, Bbuf + r * LDSS + c);
    }
}

__global__ __launch_bounds__(256) void sae_encode_gemm(
    const unsigned short* __restrict__ Xb,   // [NROWS][D_IN] bf16
    const unsigned short* __restrict__ Wb,   // [NUM_LATENTS][D_IN] bf16
    const float* __restrict__ b_enc,         // [NUM_LATENTS]
    float* __restrict__ pre)                 // [NROWS][NUM_LATENTS]
{
    __shared__ unsigned short As[2][BM * LDSS];
    __shared__ unsigned short Bs[2][BN * LDSS];

    const int tid  = threadIdx.x;
    const int lane = tid & 31;
    const int wave = tid >> 5;
    const int wm   = wave & 1;    // 0..1 -> 64-row strip of M
    const int wn   = wave >> 1;   // 0..3 -> 64-col strip of N
    const int half = lane >> 4;   // ISA lane-half for frag layout
    const int l16  = lane & 15;

    const int block_m = blockIdx.y * BM;
    const int block_n = blockIdx.x * BN;

    v8f acc[4][4];
    #pragma unroll
    for (int i = 0; i < 4; ++i)
        #pragma unroll
        for (int j = 0; j < 4; ++j)
            acc[i][j] = (v8f){0.f,0.f,0.f,0.f,0.f,0.f,0.f,0.f};

    unsigned short* As0 = &As[0][0];
    unsigned short* Bs0 = &Bs[0][0];
    const unsigned aswap = BM * LDSS;   // ushorts between the two A buffers
    const unsigned bswap = BN * LDSS;

    // prologue: fill buffer 0
    load_tiles_async(Xb, Wb, As0, Bs0, tid, block_m, block_n, 0);

    const int NK = D_IN / BK;   // 64 K-steps
    #pragma unroll 1
    for (int ks = 0; ks < NK; ++ks) {
        const unsigned aoff = (ks & 1) ? aswap : 0u;   // buffer being consumed
        const unsigned boff = (ks & 1) ? bswap : 0u;
        if (ks + 1 < NK) {
            // safe: trailing barrier of previous iteration ensured everyone is
            // done reading the other buffer before we overwrite it
            load_tiles_async(Xb, Wb, As0 + (aswap - aoff), Bs0 + (bswap - boff),
                             tid, block_m, block_n, (ks + 1) * BK);
            __builtin_amdgcn_s_wait_asynccnt(6);  // batch for current buffer done (in-order)
        } else {
            __builtin_amdgcn_s_wait_asynccnt(0);
        }
        __syncthreads();   // all waves' writes to current buffer visible

        // A frags: lanes 0-15 hold K 0-7 & 16-23; lanes 16-31 hold K 8-15 & 24-31.
        FragBF16 a[4];
        #pragma unroll
        for (int i = 0; i < 4; ++i) {
            const unsigned short* rowp = As0 + aoff + (wm * 64 + i * 16 + l16) * LDSS;
            a[i].q[0] = *(const uint4*)(rowp + half * 8);
            a[i].q[1] = *(const uint4*)(rowp + 16 + half * 8);
        }
        // B frags: one live at a time; lanes 0-15 hold K 0-7 & 16-23; lanes 16-31 hold K 8-15 & 24-31.
        #pragma unroll
        for (int j = 0; j < 4; ++j) {
            FragBF16 b;
            const unsigned short* rowp = Bs0 + boff + (wn * 64 + j * 16 + l16) * LDSS;
            b.q[0] = *(const uint4*)(rowp + half * 8);
            b.q[1] = *(const uint4*)(rowp + 16 + half * 8);
            #pragma unroll
            for (int i = 0; i < 4; ++i)
                acc[i][j] = __builtin_amdgcn_wmma_f32_16x16x32_bf16(
                    false, a[i].v, false, b.v, (short)0, acc[i][j], false, false);
        }

        __syncthreads();   // all waves done reading current buffer
    }

    // C/D layout: vgpr r, lanes 0-15 -> (M=r, N=lane); lanes 16-31 -> (M=r+8, N=lane-16).
    #pragma unroll
    for (int i = 0; i < 4; ++i)
        #pragma unroll
        for (int j = 0; j < 4; ++j) {
            int row0 = block_m + wm * 64 + i * 16 + half * 8;
            int col  = block_n + wn * 64 + j * 16 + l16;
            float bias = b_enc[col];
            #pragma unroll
            for (int r = 0; r < 8; ++r)
                pre[(size_t)(row0 + r) * NUM_LATENTS + col] = acc[i][j][r] + bias;
        }
}

// ---------------- Top-K (radix select over order-preserving key) ----------------
__device__ __forceinline__ unsigned int fkey(float f) {
    unsigned int u = __float_as_uint(f);
    return u ^ (((unsigned int)((int)u >> 31)) | 0x80000000u);
}

__global__ __launch_bounds__(256) void sae_topk(const float* __restrict__ pre,
                                                float* __restrict__ vals,
                                                int* __restrict__ idxo)
{
    __shared__ unsigned int hist[256];
    __shared__ unsigned int sh_prefix, sh_remaining, sh_gt, sh_eq;

    const int n = blockIdx.x;
    const int t = threadIdx.x;
    const float* row = pre + (size_t)n * NUM_LATENTS;

    if (t == 0) { sh_prefix = 0u; sh_remaining = TOPK; }
    unsigned int himask = 0u;
    __syncthreads();

    for (int shift = 24; shift >= 0; shift -= 8) {
        hist[t] = 0u;
        __syncthreads();
        unsigned int pfx = sh_prefix;
        for (int i = t; i < NUM_LATENTS; i += 256) {
            unsigned int k = fkey(row[i]);
            if ((k & himask) == pfx)
                atomicAdd(&hist[(k >> shift) & 255u], 1u);
        }
        __syncthreads();
        if (t == 0) {
            unsigned int rem = sh_remaining, cum = 0u;
            int b = 255;
            for (; b > 0; --b) {
                unsigned int c = hist[b];
                if (cum + c >= rem) break;
                cum += c;
            }
            sh_prefix    = pfx | ((unsigned int)b << shift);
            sh_remaining = rem - cum;
        }
        __syncthreads();
        himask |= (255u << shift);
    }

    const unsigned int T       = sh_prefix;     // exact 64th-largest key
    const unsigned int need_eq = sh_remaining;  // ties to take at threshold
    if (t == 0) { sh_gt = 0u; sh_eq = 0u; }
    __syncthreads();

    float* vrow = vals + n * TOPK;
    int*   irow = idxo + n * TOPK;
    const unsigned int gt_slots = TOPK - need_eq;
    for (int i = t; i < NUM_LATENTS; i += 256) {
        float v = row[i];
        unsigned int k = fkey(v);
        if (k > T) {
            unsigned int p = atomicAdd(&sh_gt, 1u);
            vrow[p] = v; irow[p] = i;
        } else if (k == T) {
            unsigned int p = atomicAdd(&sh_eq, 1u);
            if (p < need_eq) { vrow[gt_slots + p] = v; irow[gt_slots + p] = i; }
        }
    }
}

// ---------------- Sparse decode: y[n] = b_dec + sum_k vals[n,k]*W_dec[idx[n,k]] ----------------
__global__ __launch_bounds__(256) void sae_decode(const float* __restrict__ vals,
                                                  const int* __restrict__ idx,
                                                  const float* __restrict__ Wdec,
                                                  const float* __restrict__ bdec,
                                                  float* __restrict__ y)
{
    const int n = blockIdx.x;
    const int t = threadIdx.x;
    const float4* bd = (const float4*)bdec;
    float4 a0 = bd[t];
    float4 a1 = bd[t + 256];
    const float* vr = vals + n * TOPK;
    const int*   ir = idx  + n * TOPK;
    #pragma unroll 4
    for (int k = 0; k < TOPK; ++k) {
        float a = vr[k];
        int   l = ir[k];
        const float4* w = (const float4*)(Wdec + (size_t)l * D_IN);
        float4 w0 = w[t], w1 = w[t + 256];
        a0.x = fmaf(a, w0.x, a0.x); a0.y = fmaf(a, w0.y, a0.y);
        a0.z = fmaf(a, w0.z, a0.z); a0.w = fmaf(a, w0.w, a0.w);
        a1.x = fmaf(a, w1.x, a1.x); a1.y = fmaf(a, w1.y, a1.y);
        a1.z = fmaf(a, w1.z, a1.z); a1.w = fmaf(a, w1.w, a1.w);
    }
    float4* yo = (float4*)(y + (size_t)n * D_IN);
    yo[t] = a0; yo[t + 256] = a1;
}

// ---------------- launch ----------------
extern "C" void kernel_launch(void* const* d_in, const int* in_sizes, int n_in,
                              void* d_out, int out_size, void* d_ws, size_t ws_size,
                              hipStream_t stream) {
    const float* x     = (const float*)d_in[0];
    const float* W_enc = (const float*)d_in[1];
    const float* b_enc = (const float*)d_in[2];
    const float* W_dec = (const float*)d_in[3];
    const float* b_dec = (const float*)d_in[4];
    float* y = (float*)d_out;

    char* p = (char*)d_ws;
    float*          pre  = (float*)p;          p += (size_t)NROWS * NUM_LATENTS * 4;
    unsigned short* Xb   = (unsigned short*)p; p += (size_t)NROWS * D_IN * 2;
    unsigned short* Wb   = (unsigned short*)p; p += (size_t)NUM_LATENTS * D_IN * 2;
    float*          vals = (float*)p;          p += (size_t)NROWS * TOPK * 4;
    int*            idx  = (int*)p;

    // 1) convert to bf16
    {
        int n4x = (NROWS * D_IN) / 4;
        cvt_f32_bf16<<<n4x / 256, 256, 0, stream>>>(x, Xb, n4x);
        int n4w = (NUM_LATENTS * D_IN) / 4;
        cvt_f32_bf16<<<n4w / 256, 256, 0, stream>>>(W_enc, Wb, n4w);
    }
    // 2) encoder GEMM (bf16 WMMA, f32 accumulate, async double-buffered LDS)
    {
        dim3 grid(NUM_LATENTS / BN, NROWS / BM);
        sae_encode_gemm<<<grid, 256, 0, stream>>>(Xb, Wb, b_enc, pre);
    }
    // 3) top-64 per row
    sae_topk<<<NROWS, 256, 0, stream>>>(pre, vals, idx);
    // 4) sparse decode
    sae_decode<<<NROWS, 256, 0, stream>>>(vals, idx, W_dec, b_dec, y);
}
